// Q_AvgPoolBranch_73581379715317
// MI455X (gfx1250) — 8-bit matrix operand order repaired and checked by reading, compiled, not run on hardware
//
#include <hip/hip_runtime.h>
#include <hip/hip_bf16.h>

typedef __attribute__((ext_vector_type(8))) int v8i_t;

#define B_    128
#define CIN   768
#define COUT  192
#define HH    17
#define WW    17
#define HWS   289
#define NPOS  (B_*HWS)
#define NX    (B_*CIN*HWS)
#define NOUT  (B_*COUT*HWS)
#define QN    127.0f

#define WS_SLOTS 0
#define WS_BINT  256
#define WS_SCALE 1024
#define WS_WQ    2048
#define WS_XQ    149504
#define WS_XI    28559360

__device__ inline unsigned blockReduceMaxU(unsigned v) {
    __shared__ unsigned sm[256];
    sm[threadIdx.x] = v;
    __syncthreads();
    for (int s = 128; s > 0; s >>= 1) {
        if ((int)threadIdx.x < s) {
            unsigned o = sm[threadIdx.x + s];
            if (o > sm[threadIdx.x]) sm[threadIdx.x] = o;
        }
        __syncthreads();
    }
    return sm[0];
}

__global__ void k_init(unsigned* slots) {
    if (threadIdx.x < 16) slots[threadIdx.x] = 0u;
}

__global__ void k_absmax_x(const float* __restrict__ x, unsigned* __restrict__ slots) {
    int i = blockIdx.x * 256 + threadIdx.x;
    float a = 0.0f;
    if (i < NX) a = fabsf(x[i]);
    unsigned m = blockReduceMaxU(__float_as_uint(a));
    if (threadIdx.x == 0) atomicMax(&slots[0], m);
}

__global__ void k_quant_x(const float* __restrict__ x, signed char* __restrict__ xq,
                          const unsigned* __restrict__ slots) {
    int i = blockIdx.x * 256 + threadIdx.x;
    if (i >= NX) return;
    float s0 = fmaxf(__uint_as_float(slots[0]) / QN, 1e-8f);
    float q = rintf(x[i] / s0);
    q = fminf(fmaxf(q, -128.0f), 127.0f);
    xq[i] = (signed char)(int)q;
}

__global__ void k_pool_max(const signed char* __restrict__ xq, unsigned* __restrict__ slots) {
    int i = blockIdx.x * 256 + threadIdx.x;
    int asum = 0;
    if (i < NX) {
        int hw = i % HWS; int rest = i / HWS;
        int c = rest % CIN; int b = rest / CIN;
        int h = hw / WW, w = hw % WW;
        const signed char* p = xq + (size_t)(b * CIN + c) * HWS;
        int sum = 0;
        #pragma unroll
        for (int dh = -1; dh <= 1; ++dh) {
            int hh = h + dh;
            if (hh < 0 || hh >= HH) continue;
            #pragma unroll
            for (int dw = -1; dw <= 1; ++dw) {
                int ww2 = w + dw;
                if (ww2 < 0 || ww2 >= WW) continue;
                sum += (int)p[hh * WW + ww2];
            }
        }
        asum = sum < 0 ? -sum : sum;
    }
    unsigned m = blockReduceMaxU((unsigned)asum);
    if (threadIdx.x == 0) atomicMax(&slots[1], m);
}

__global__ void k_pool_xi(const signed char* __restrict__ xq, signed char* __restrict__ xi,
                          const unsigned* __restrict__ slots) {
    int i = blockIdx.x * 256 + threadIdx.x;
    if (i >= NX) return;
    int c = i % CIN; int rest = i / CIN;
    int hw = rest % HWS; int b = rest / HWS;
    int h = hw / WW, w = hw % WW;
    const signed char* p = xq + (size_t)(b * CIN + c) * HWS;
    int sum = 0;
    #pragma unroll
    for (int dh = -1; dh <= 1; ++dh) {
        int hh = h + dh;
        if (hh < 0 || hh >= HH) continue;
        #pragma unroll
        for (int dw = -1; dw <= 1; ++dw) {
            int ww2 = w + dw;
            if (ww2 < 0 || ww2 >= WW) continue;
            sum += (int)p[hh * WW + ww2];
        }
    }
    float s0 = fmaxf(__uint_as_float(slots[0]) / QN, 1e-8f);
    float x2max = ((float)slots[1] / 9.0f) * s0;
    float s1 = fmaxf(x2max / QN, 1e-8f);
    float x2 = ((float)sum / 9.0f) * s0;
    float q = rintf(x2 / s1);
    q = fminf(fmaxf(q, -128.0f), 127.0f);
    xi[(size_t)(b * HWS + hw) * CIN + c] = (signed char)(int)q;
}

__global__ void k_quant_w(const float* __restrict__ cw, const float* __restrict__ gamma,
                          const float* __restrict__ beta, const float* __restrict__ mean,
                          const float* __restrict__ var, signed char* __restrict__ wq,
                          float* __restrict__ bint, float* __restrict__ scale,
                          const unsigned* __restrict__ slots) {
    int o = threadIdx.x;
    if (o >= COUT) return;
    float s0 = fmaxf(__uint_as_float(slots[0]) / QN, 1e-8f);
    float x2max = ((float)slots[1] / 9.0f) * s0;
    float s1 = fmaxf(x2max / QN, 1e-8f);
    float g = gamma[o];
    float stdv = sqrtf(var[o] + 1e-5f);
    float sc = g / stdv;
    const float* wr = cw + (size_t)o * CIN;
    float mx = 0.0f;
    for (int c = 0; c < CIN; ++c) mx = fmaxf(mx, fabsf(wr[c] * sc));
    float w_sf = fmaxf(mx / QN, 1e-8f);
    signed char* wo = wq + (size_t)o * CIN;
    for (int c = 0; c < CIN; ++c) {
        float q = rintf(wr[c] * sc / w_sf);
        q = fminf(fmaxf(q, -128.0f), 127.0f);
        wo[c] = (signed char)(int)q;
    }
    float b_fold = beta[o] - g * mean[o] / stdv;
    bint[o] = rintf(b_fold / (w_sf * s1));
    scale[o] = w_sf * s1;
}

#define N2TILES (COUT / 32)
#define WAVEJOBS ((NPOS / 64) * N2TILES)
__global__ void __launch_bounds__(256)
k_gemm(const signed char* __restrict__ xi, const signed char* __restrict__ wq,
       const float* __restrict__ bint, const float* __restrict__ scale,
       float* __restrict__ out, unsigned* __restrict__ slots) {
    const int wave = (blockIdx.x * 256 + threadIdx.x) >> 5;
    if (wave >= WAVEJOBS) return;
    const int lane = threadIdx.x & 31;
    const int m4 = wave / N2TILES;
    const int n2 = wave % N2TILES;
    const int Mbase = m4 * 64;
    const int Nbase = n2 * 32;
    const int l16 = lane & 15;
    const int half = lane >> 4;

    v8i_t acc[4][2];
    #pragma unroll
    for (int t = 0; t < 4; ++t)
        #pragma unroll
        for (int j = 0; j < 2; ++j)
            acc[t][j] = (v8i_t){0,0,0,0,0,0,0,0};

    const signed char* wrow0 = wq + (size_t)(Nbase + l16) * CIN + half * 16;
    const signed char* wrow1 = wq + (size_t)(Nbase + 16 + l16) * CIN + half * 16;
    const signed char* arow = xi + (size_t)(Mbase + l16) * CIN + half * 16;

    #pragma unroll 2
    for (int k0 = 0; k0 < CIN; k0 += 64) {
        int4 b00 = *reinterpret_cast<const int4*>(wrow0 + k0);
        int4 b01 = *reinterpret_cast<const int4*>(wrow0 + k0 + 32);
        int4 b10 = *reinterpret_cast<const int4*>(wrow1 + k0);
        int4 b11 = *reinterpret_cast<const int4*>(wrow1 + k0 + 32);
        v8i_t Bm0 = { b00.x, b00.y, b00.z, b00.w, b01.x, b01.y, b01.z, b01.w };
        v8i_t Bm1 = { b10.x, b10.y, b10.z, b10.w, b11.x, b11.y, b11.z, b11.w };
        #pragma unroll
        for (int t = 0; t < 4; ++t) {
            const signed char* ap = arow + (size_t)t * 16 * CIN + k0;
            int4 g0 = *reinterpret_cast<const int4*>(ap);
            int4 g1 = *reinterpret_cast<const int4*>(ap + 32);
            v8i_t Am = { g0.x, g0.y, g0.z, g0.w, g1.x, g1.y, g1.z, g1.w };
            acc[t][0] = __builtin_amdgcn_wmma_i32_16x16x64_iu8(
                true, Am, true, Bm0, acc[t][0], false, false);
            acc[t][1] = __builtin_amdgcn_wmma_i32_16x16x64_iu8(
                true, Am, true, Bm1, acc[t][1], false, false);
        }
    }

    float lmax = 0.0f;
    #pragma unroll
    for (int j = 0; j < 2; ++j) {
        const int o = Nbase + j * 16 + l16;
        const float bo = bint[o];
        const float so = scale[o];
        float* outc = out + (size_t)o * HWS;
        #pragma unroll
        for (int t = 0; t < 4; ++t) {
            int pos0 = Mbase + t * 16 + half * 8;
            int b = pos0 / HWS;
            int hw = pos0 - b * HWS;
            #pragma unroll
            for (int v = 0; v < 8; ++v) {
                float f = ((float)acc[t][j][v] + bo) * so;
                f = f > 0.0f ? f : 0.0f;
                outc[(size_t)b * (COUT * HWS) + hw] = f;
                lmax = fmaxf(lmax, f);
                if (++hw == HWS) { hw = 0; ++b; }
            }
        }
    }
    atomicMax(&slots[2], __float_as_uint(lmax));
}

__global__ void k_requant(float* __restrict__ out, const unsigned* __restrict__ slots) {
    int i = blockIdx.x * 256 + threadIdx.x;
    float amax = __uint_as_float(slots[2]);
    float s2 = fmaxf(amax / QN, 1e-8f);
    if (i < NOUT) {
        float q = rintf(out[i] / s2);
        q = fminf(fmaxf(q, -128.0f), 127.0f);
        out[i] = q * s2;
    }
    if (i == 0) out[NOUT] = s2;
}

extern "C" void kernel_launch(void* const* d_in, const int* in_sizes, int n_in,
                              void* d_out, int out_size, void* d_ws, size_t ws_size,
                              hipStream_t stream) {
    const float* x     = (const float*)d_in[0];
    const float* cw    = (const float*)d_in[1];
    const float* gamma = (const float*)d_in[2];
    const float* beta  = (const float*)d_in[3];
    const float* mean  = (const float*)d_in[4];
    const float* var   = (const float*)d_in[5];
    float* out = (float*)d_out;

    char* ws = (char*)d_ws;
    unsigned* slots   = (unsigned*)(ws + WS_SLOTS);
    float* bint       = (float*)(ws + WS_BINT);
    float* scale      = (float*)(ws + WS_SCALE);
    signed char* wq   = (signed char*)(ws + WS_WQ);
    signed char* xq   = (signed char*)(ws + WS_XQ);
    signed char* xi   = (signed char*)(ws + WS_XI);

    const int bx = (NX + 255) / 256;

    k_init<<<1, 64, 0, stream>>>(slots);
    k_absmax_x<<<bx, 256, 0, stream>>>(x, slots);
    k_quant_x<<<bx, 256, 0, stream>>>(x, xq, slots);
    k_pool_max<<<bx, 256, 0, stream>>>(xq, slots);
    k_pool_xi<<<bx, 256, 0, stream>>>(xq, xi, slots);
    k_quant_w<<<1, 256, 0, stream>>>(cw, gamma, beta, mean, var, wq, bint, scale, slots);

    k_gemm<<<(WAVEJOBS + 7) / 8, 256, 0, stream>>>(xi, wq, bint, scale, out, slots);

    k_requant<<<(NOUT + 255) / 256, 256, 0, stream>>>(out, slots);
}
